// GATBlock_24532853195196
// MI455X (gfx1250) — matrix operand order repaired and checked by reading, compiled, not run on hardware
//
#include <hip/hip_runtime.h>
#include <hip/hip_bf16.h>
#include <cstdint>
#include <cstddef>

// ---------------------------------------------------------------------------
// GAT block for MI455X (gfx1250, wave32, WMMA).
// N=6144 nodes, C=CO=128, K=2 heads. adj is 151MB -> memory roofline ~6.5us
// per pass @23.3TB/s; atten@h is 19.3 GFLOP -> bf16 WMMA with f32 accum.
// adj is read exactly twice (stats pass + accumulate pass); e/atten are never
// materialized. s_dst is staged to LDS via CDNA5 async-to-LDS loads.
// ---------------------------------------------------------------------------

#define NN   6144      // nodes
#define CD   128       // feature dim (C == CO)
#define KH   2         // heads
#define JS   4         // j-range splits in attention kernel
#define JCH  (NN / JS) // 1536 j's per wave chunk

typedef __attribute__((ext_vector_type(16))) __bf16 v16bf;
typedef __attribute__((ext_vector_type(8)))  __bf16 v8bf;
typedef __attribute__((ext_vector_type(8)))  float  v8f;

static __device__ __forceinline__ unsigned short f2bf(float f) {
    union { float f; unsigned u; } v; v.f = f;
    unsigned r = v.u + 0x7FFFu + ((v.u >> 16) & 1u);   // round-to-nearest-even
    return (unsigned short)(r >> 16);
}

// ---------------------------------------------------------------- k0: W0->bf16
__global__ void k0_cvt_w0(const float* __restrict__ w, unsigned short* __restrict__ o, int n) {
    int t = blockIdx.x * blockDim.x + threadIdx.x;
    if (t < n) o[t] = f2bf(w[t]);
}

// ------------------------------------------------------------- k1: LN(x)->bf16
__global__ void k1_layernorm(const float* __restrict__ x, const float* __restrict__ g,
                             const float* __restrict__ b, unsigned short* __restrict__ xn) {
    int row = blockIdx.x, c = threadIdx.x;
    int lane = c & 31, w = c >> 5;
    float v = x[(size_t)row * CD + c];
    float s = v, s2 = v * v;
    #pragma unroll
    for (int o = 16; o; o >>= 1) { s += __shfl_xor(s, o, 32); s2 += __shfl_xor(s2, o, 32); }
    __shared__ float t1[4], t2[4];
    if (lane == 0) { t1[w] = s; t2[w] = s2; }
    __syncthreads();
    float tot  = t1[0] + t1[1] + t1[2] + t1[3];
    float tot2 = t2[0] + t2[1] + t2[2] + t2[3];
    float mean = tot * (1.0f / CD);
    float var  = tot2 * (1.0f / CD) - mean * mean;
    float inv  = rsqrtf(var + 1e-5f);
    xn[(size_t)row * CD + c] = f2bf((v - mean) * inv * g[c] + b[c]);
}

// ---------------------------------------------- k2: h = LN(x) @ W0^T  (WMMA)
// one wave per (16-row tile, 16-col tile, head); D stored transposed as bf16
// hT[head][co][i] so k5's B-matrix loads are two 16B v8bf runs along i.
__global__ void k2_h_gemm(const unsigned short* __restrict__ xn_u,
                          const unsigned short* __restrict__ w0_u,
                          unsigned short* __restrict__ hT_u) {
    const __bf16* xn = (const __bf16*)xn_u;
    const __bf16* w0 = (const __bf16*)w0_u;
    int wid  = blockIdx.x * 8 + (threadIdx.x >> 5);
    int lane = threadIdx.x & 31;
    int head = wid & 1, cot = (wid >> 1) & 7, rt = wid >> 4;
    int i0   = rt * 16;
    int row  = i0 + (lane & 15);
    int co   = cot * 16 + (lane & 15);
    int off0 = (lane < 16) ? 0 : 8;    // A-matrix K sublane offset
    int offB = (lane < 16) ? 0 : 16;   // unused; B loads use off0 like A

    v8f acc;
    #pragma unroll
    for (int v = 0; v < 8; ++v) acc[v] = 0.0f;

    const __bf16* arow = xn + (size_t)row * CD;
    const __bf16* brow = w0 + ((size_t)head * CD + co) * CD;   // W0[head][co][*]
    #pragma unroll
    for (int c = 0; c < CD; c += 32) {
        v8bf lo = *(const v8bf*)(arow + c + off0);
        v8bf hi = *(const v8bf*)(arow + c + off0 + 16);
        v16bf A = __builtin_shufflevector(lo, hi, 0,1,2,3,4,5,6,7,8,9,10,11,12,13,14,15);
        v8bf blo = *(const v8bf*)(brow + c + off0);
        v8bf bhi = *(const v8bf*)(brow + c + off0 + 16);
        v16bf B = __builtin_shufflevector(blo, bhi, 0,1,2,3,4,5,6,7,8,9,10,11,12,13,14,15);
        acc = __builtin_amdgcn_wmma_f32_16x16x32_bf16(false, A, false, B,
                                                      (short)0, acc, false, false);
    }
    // D layout: lane = col co, VGPR v = row (v + 8*(lane>>4)) -> 8 consecutive i
    int ib = i0 + 8 * (lane >> 4);
    __bf16* dst = (__bf16*)hT_u + ((size_t)head * CD + co) * NN + ib;
    v8bf vb;
    #pragma unroll
    for (int v = 0; v < 8; ++v) vb[v] = (__bf16)acc[v];   // packed bf16 converts
    *(v8bf*)dst = vb;                                     // single 16B store
}

// -------------------------------------------- k3: s_src/s_dst = h . attn parts
__global__ void k3_scores(const unsigned short* __restrict__ hT_u,
                          const float* __restrict__ aw,
                          float* __restrict__ ss, float* __restrict__ sd) {
    int i = blockIdx.x * blockDim.x + threadIdx.x;
    if (i >= NN) return;
    const __bf16* hT = (const __bf16*)hT_u;
    #pragma unroll
    for (int h = 0; h < KH; ++h) {
        const float* a = aw + (size_t)h * 2 * CD;
        float s0 = 0.f, s1 = 0.f;
        for (int c = 0; c < CD; ++c) {
            float hv = (float)hT[((size_t)h * CD + c) * NN + i];  // coalesced over i
            s0 += hv * a[c];
            s1 += hv * a[CD + c];
        }
        ss[h * NN + i] = s0;
        sd[h * NN + i] = s1;
    }
}

// ----------------------------- k4: per-row online softmax stats, both heads,
// one adj read shared. One wave per row.
__global__ void k4_stats(const float* __restrict__ adj, const float* __restrict__ ss,
                         const float* __restrict__ sd, float* __restrict__ mo,
                         float* __restrict__ lo) {
    int i    = blockIdx.x * 8 + (threadIdx.x >> 5);
    int lane = threadIdx.x & 31;
    float ssrc0 = ss[i], ssrc1 = ss[NN + i];
    float m0 = -3.0e38f, l0 = 0.f, m1 = -3.0e38f, l1 = 0.f;
    const float* arow = adj + (size_t)i * NN;
    for (int it = 0; it < NN / 128; ++it) {
        int j = (it * 32 + lane) * 4;
        __builtin_prefetch(arow + j + 128, 0, 0);        // next block of the stream
        float4 av = *(const float4*)(arow + j);
        float4 d0 = *(const float4*)(sd + j);
        float4 d1 = *(const float4*)(sd + NN + j);
        float aa[4]  = {av.x, av.y, av.z, av.w};
        float dd0[4] = {d0.x, d0.y, d0.z, d0.w};
        float dd1[4] = {d1.x, d1.y, d1.z, d1.w};
        #pragma unroll
        for (int t = 0; t < 4; ++t) {
            bool un = (aa[t] > 0.f) || (j + t == i);
            if (un) {
                float s = ssrc0 + dd0[t];
                float e = fmaxf(s, 0.01f * s);                 // leaky_relu
                if (e <= m0) l0 += __expf(e - m0);
                else { l0 = l0 * __expf(m0 - e) + 1.f; m0 = e; }
                s = ssrc1 + dd1[t];
                e = fmaxf(s, 0.01f * s);
                if (e <= m1) l1 += __expf(e - m1);
                else { l1 = l1 * __expf(m1 - e) + 1.f; m1 = e; }
            }
        }
    }
    #pragma unroll
    for (int o = 16; o; o >>= 1) {
        float mx = __shfl_xor(m0, o, 32), lx = __shfl_xor(l0, o, 32);
        float mn = fmaxf(m0, mx);
        l0 = l0 * __expf(m0 - mn) + lx * __expf(mx - mn); m0 = mn;
        mx = __shfl_xor(m1, o, 32); lx = __shfl_xor(l1, o, 32);
        mn = fmaxf(m1, mx);
        l1 = l1 * __expf(m1 - mn) + lx * __expf(mx - mn); m1 = mn;
    }
    if (lane == 0) { mo[i] = m0; lo[i] = l0; mo[NN + i] = m1; lo[NN + i] = l1; }
}

// --------------------------------- k5: o = exp(e - m) @ h  (bf16 WMMA, fused)
// wave = (16-row tile, head, j-quarter). A built per lane in the ISA 16-bit
// A-matrix layout; B = hT slice (two 16B v8bf loads, 32B apart). 8 c-tiles -> 64 acc.
// Per-wave s_dst chunk (1536 f32) staged to LDS once via async-to-LDS loads.
__global__ void k5_attn(const float* __restrict__ adj, const unsigned short* __restrict__ hT_u,
                        const float* __restrict__ ss, const float* __restrict__ sd,
                        const float* __restrict__ mrow, float* __restrict__ opart) {
    __shared__ __align__(16) float sdbuf[8 * JCH];    // 48 KB, one slot per wave
    int wid  = blockIdx.x * 8 + (threadIdx.x >> 5);
    int lane = threadIdx.x & 31;
    int w    = threadIdx.x >> 5;
    int head = wid & 1, js = (wid >> 1) & 3, rt = wid >> 3;
    int i0   = rt * 16;
    int i    = i0 + (lane & 15);
    int off0 = (lane < 16) ? 0 : 8;
    int jhalf = (lane < 16) ? 0 : 16;
    int cB   = lane & 15;
    const int jstart = js * JCH;

    // ---- async stage of this wave's s_dst chunk into LDS (ASYNCcnt path) ----
    {
        const float* chunk = sd + (size_t)head * NN + jstart;
        uint32_t ldsbase = (uint32_t)(size_t)&sdbuf[w * JCH];   // low 32b == LDS offset
        #pragma unroll
        for (int it = 0; it < JCH / 128; ++it) {                // 12 x b128 per lane
            uint32_t dst = ldsbase + (uint32_t)(it * 128 + lane * 4) * 4u;
            unsigned long long src = (unsigned long long)(const void*)(chunk + it * 128 + lane * 4);
            asm volatile("global_load_async_to_lds_b128 %0, %1, off"
                         :: "v"(dst), "v"(src) : "memory");
        }
        asm volatile("s_wait_asynccnt 0" ::: "memory");
    }
    const float* sdl = &sdbuf[w * JCH] - jstart;   // indexable by absolute j

    const float L2E = 1.4426950408889634f;
    float ssrc = ss[head * NN + i];
    float mi2  = mrow[head * NN + i] * L2E;        // hoisted: exp(e-m)=exp2(e*L2E - m*L2E)
    const float*  arow = adj + (size_t)i * NN;
    const __bf16* hT   = (const __bf16*)hT_u + (size_t)head * CD * NN;

    v8f acc[8];
    #pragma unroll
    for (int ct = 0; ct < 8; ++ct)
        #pragma unroll
        for (int v = 0; v < 8; ++v) acc[ct][v] = 0.0f;

    for (int jb = jstart; jb < jstart + JCH; jb += 32) {
        __builtin_prefetch(arow + jb + 32 + off0, 0, 0);        // stream adj ahead
        float4 a0 = *(const float4*)(arow + jb + off0);
        float4 a1 = *(const float4*)(arow + jb + off0 + 4);
        float4 a2 = *(const float4*)(arow + jb + 16 + off0);
        float4 a3 = *(const float4*)(arow + jb + 16 + off0 + 4);
        float4 d0 = *(const float4*)(sdl + jb + off0);          // ds_load_b128
        float4 d1 = *(const float4*)(sdl + jb + off0 + 4);
        float4 d2 = *(const float4*)(sdl + jb + 16 + off0);
        float4 d3 = *(const float4*)(sdl + jb + 16 + off0 + 4);
        float av[16] = {a0.x,a0.y,a0.z,a0.w, a1.x,a1.y,a1.z,a1.w,
                        a2.x,a2.y,a2.z,a2.w, a3.x,a3.y,a3.z,a3.w};
        float dv[16] = {d0.x,d0.y,d0.z,d0.w, d1.x,d1.y,d1.z,d1.w,
                        d2.x,d2.y,d2.z,d2.w, d3.x,d3.y,d3.z,d3.w};
        v16bf A;
        #pragma unroll
        for (int t = 0; t < 16; ++t) {
            int j = jb + off0 + ((t < 8) ? t : (t - 8 + 16));
            bool un = (av[t] > 0.f) || (j == i);
            float s = ssrc + dv[t];
            float e = fmaxf(s, 0.01f * s);                      // leaky_relu
            float p = un ? exp2f(fmaf(e, L2E, -mi2)) : 0.0f;    // masked -> exact 0
            A[t] = (__bf16)p;
        }
        #pragma unroll
        for (int ct = 0; ct < 8; ++ct) {
            const __bf16* bp = hT + (size_t)(ct * 16 + cB) * NN + jb + off0;
            v8bf blo = *(const v8bf*)(bp);
            v8bf bhi = *(const v8bf*)(bp + 16);
            v16bf B = __builtin_shufflevector(blo, bhi, 0,1,2,3,4,5,6,7,8,9,10,11,12,13,14,15);
            acc[ct] = __builtin_amdgcn_wmma_f32_16x16x32_bf16(false, A, false, B,
                                                              (short)0, acc[ct], false, false);
        }
    }
    // deterministic partial store (no atomics): opart[js][head][i][c]
    float* op = opart + (((size_t)js * KH + head) * NN + i0) * CD;
    int ib = 8 * (lane >> 4);
    #pragma unroll
    for (int ct = 0; ct < 8; ++ct)
        #pragma unroll
        for (int v = 0; v < 8; ++v)
            op[(size_t)(ib + v) * CD + ct * 16 + (lane & 15)] = acc[ct][v];
}

// --------- k6: sum partials, /l, +x, LN(g2,b2), @ff1^T, ELU, average heads
__global__ void k6_epilogue(const float* __restrict__ x, const float* __restrict__ opart,
                            const float* __restrict__ lrow, const float* __restrict__ g2,
                            const float* __restrict__ b2, const float* __restrict__ ff1,
                            float* __restrict__ out) {
    int i = blockIdx.x, c = threadIdx.x;
    int lane = c & 31, w = c >> 5;
    __shared__ float t1[4], t2[4];
    __shared__ float sh_t[KH][CD];
    #pragma unroll
    for (int h = 0; h < KH; ++h) {
        float o = 0.f;
        #pragma unroll
        for (int js = 0; js < JS; ++js)
            o += opart[(((size_t)js * KH + h) * NN + i) * CD + c];
        float t = o / lrow[h * NN + i] + x[(size_t)i * CD + c];
        float s = t, s2 = t * t;
        #pragma unroll
        for (int of = 16; of; of >>= 1) { s += __shfl_xor(s, of, 32); s2 += __shfl_xor(s2, of, 32); }
        if (lane == 0) { t1[w] = s; t2[w] = s2; }
        __syncthreads();
        float tot  = t1[0] + t1[1] + t1[2] + t1[3];
        float tot2 = t2[0] + t2[1] + t2[2] + t2[3];
        float mean = tot * (1.0f / CD);
        float var  = tot2 * (1.0f / CD) - mean * mean;
        sh_t[h][c] = (t - mean) * rsqrtf(var + 1e-5f) * g2[c] + b2[c];
        __syncthreads();   // t1/t2 reused next head
    }
    float res = 0.f;
    #pragma unroll
    for (int h = 0; h < KH; ++h) {
        const float* wr = ff1 + ((size_t)h * CD + c) * CD;   // ff1[h][c][*]
        float a = 0.f;
        #pragma unroll 4
        for (int c2 = 0; c2 < CD; ++c2) a += sh_t[h][c2] * wr[c2];
        float e = (a > 0.f) ? a : (__expf(a) - 1.0f);        // ELU
        res += 0.5f * e;                                     // / K
    }
    out[(size_t)i * CD + c] = res;
}

// ---------------------------------------------------------------------------
extern "C" void kernel_launch(void* const* d_in, const int* in_sizes, int n_in,
                              void* d_out, int out_size, void* d_ws, size_t ws_size,
                              hipStream_t stream) {
    const float* x    = (const float*)d_in[0];
    const float* adj  = (const float*)d_in[1];
    const float* ff0  = (const float*)d_in[2];
    const float* ff1  = (const float*)d_in[3];
    const float* aw   = (const float*)d_in[4];
    const float* g1   = (const float*)d_in[5];
    const float* b1   = (const float*)d_in[6];
    const float* g2   = (const float*)d_in[7];
    const float* b2   = (const float*)d_in[8];
    float* out        = (float*)d_out;

    // workspace layout (all 256B aligned)
    char* ws = (char*)d_ws;
    unsigned short* w0bf = (unsigned short*)(ws);                      //  64 KB
    unsigned short* xnbf = (unsigned short*)(ws + 65536);              // 1.5 MB
    unsigned short* hTbf = (unsigned short*)(ws + 1638400);            // 3.0 MB
    float* ss            = (float*)(ws + 4784128);                     //  48 KB
    float* sd            = (float*)(ws + 4833280);                     //  48 KB
    float* mrow          = (float*)(ws + 4882432);                     //  48 KB
    float* lrow          = (float*)(ws + 4931584);                     //  48 KB
    float* opart         = (float*)(ws + 4980736);                     //  24 MB

    k0_cvt_w0   <<<128,  256, 0, stream>>>(ff0, w0bf, KH * CD * CD);
    k1_layernorm<<<NN,   128, 0, stream>>>(x, g1, b1, xnbf);
    k2_h_gemm   <<<768,  256, 0, stream>>>(xnbf, w0bf, hTbf);          // 6144 waves
    k3_scores   <<<24,   256, 0, stream>>>(hTbf, aw, ss, sd);
    k4_stats    <<<768,  256, 0, stream>>>(adj, ss, sd, mrow, lrow);   // wave/row
    k5_attn     <<<384,  256, 0, stream>>>(adj, hTbf, ss, sd, mrow, opart); // 3072 waves
    k6_epilogue <<<NN,   128, 0, stream>>>(x, opart, lrow, g2, b2, ff1, out);
}
